// PatchAttentionMask_11759620456569
// MI455X (gfx1250) — matrix operand order repaired and checked by reading, not compiled, not run on hardware
//
#include <hip/hip_runtime.h>
#include <hip/hip_bf16.h>
#include <math.h>

// ---------------------------------------------------------------------------
// PatchAttentionMask for MI455X (gfx1250, wave32).
// Roofline: dominated by one 256 MiB read of x (~11us @ 23.3 TB/s). Everything
// else is <= 1 MiB per tensor, so we fuse aggressively, use the CDNA5 WMMA
// path (v_wmma_f32_16x16x32_f16) for the channel-attention GEMMs, and the
// Tensor Data Mover (tensor_load_to_lds with LDS row padding) to build the
// zero-padded conv tile in K2.
// ---------------------------------------------------------------------------

typedef __attribute__((ext_vector_type(16))) _Float16 v16h;
typedef __attribute__((ext_vector_type(8)))  float    v8f;
typedef __attribute__((ext_vector_type(4)))  unsigned int v4u;
typedef __attribute__((ext_vector_type(8)))  int      v8i;
typedef __attribute__((ext_vector_type(4)))  int      v4i;

#define BB 16
#define CC 256
#define HH 128
#define WW 128
#define PP 16
#define LL 64           // 8x8 patches per image
#define NP 1024         // B * L
#define P2 256          // P*P
#define PLANE (HH*WW)   // 16384
#define INV_TEMP 10.0f
#define ALPHA_ 0.7f
#define TOPK 16

// workspace layout (float offsets)
#define OFF_FEA  0         // 1024*256  fea = channel-mean(x^2) + maxpool
#define OFF_CM   262144    // 1024*256  cm  = spatial-mean(x^2)
#define OFF_HPRE 524288    // 1024*256  conv1 output (pre-BN)
#define OFF_STAT 786432    // 8         BN mean/var stats (written by K2b)
#define OFF_SS   786440    // 1024      spatial scores
#define OFF_CS   787464    // 1024      channel scores
#define OFF_CMH  788488    // 1024*256 halves (f16 cm for WMMA)
#define OFF_PSUM 919560    // 1024      per-block BN partial sums (deterministic)
#define OFF_PSSQ 920584    // 1024      per-block BN partial sumsq
#define WS_NEEDED_BYTES ((size_t)(920584 + 1024) * 4)

// ---------------------------------------------------------------------------
// TDM: DMA a contiguous 16x16 f32 tile from global into LDS, padding each
// 16-dword row with 6 dwords so rows land at stride 22 floats (the interior
// of a 22x22 zero-ring conv tile). D# per ISA 8.3/8.4:
//   group0: count=1 | lds_addr | global_addr[56:0] | type=2
//   group1: data_size=2(4B), pad_enable, pad_interval=3(16DW), pad_amount=5(6DW),
//           tensor_dim0/1=16, tile_dim0/1=16, tensor_dim0_stride=16
// Issued by one wave; completion via s_wait_tensorcnt 0.
// ---------------------------------------------------------------------------
__device__ __forceinline__ void tdm_load_tile16x16_pad22(const float* gsrc, float* ldst) {
  unsigned lds_off = (unsigned)(unsigned long long)ldst;   // low 32 bits of flat = LDS offset
  unsigned long long ga = (unsigned long long)gsrc;
  v4u g0;
  g0[0] = 1u;                                          // count=1, user descriptor
  g0[1] = lds_off;                                     // lds_addr
  g0[2] = (unsigned)(ga & 0xffffffffu);                // global_addr[31:0]
  g0[3] = (unsigned)((ga >> 32) & 0x01ffffffu) | (2u << 30);  // global_addr[56:32] | type=2
  v8i g1;
  g1[0] = (2 << 16) | (1 << 20) | (3 << 22) | (5 << 25);  // data_size=4B, pad 16DW->+6DW
  g1[1] = 16 << 16;                                    // tensor_dim0 = 16 (bits 79:48)
  g1[2] = 16 << 16;                                    // tensor_dim1 = 16 (bits 111:80)
  g1[3] = 16 << 16;                                    // tile_dim0   = 16 (bits 127:112)
  g1[4] = 16;                                          // tile_dim1   = 16 (bits 143:128)
  g1[5] = 16;                                          // tensor_dim0_stride = 16
  g1[6] = 0; g1[7] = 0;
  v4i g2 = {0, 0, 0, 0};
  v4i g3 = {0, 0, 0, 0};
#if __clang_major__ >= 23
  v8i g4 = {0, 0, 0, 0, 0, 0, 0, 0};
  __builtin_amdgcn_tensor_load_to_lds(g0, g1, g2, g3, g4, 0);
#else
  __builtin_amdgcn_tensor_load_to_lds(g0, g1, g2, g3, 0);
#endif
}

// ---------------------------------------------------------------------------
// K1: one coalesced pass over x for fea (channel reduction, lane = spatial
// position) with fused 3x3 maxpool; second L2-resident pass for cm (thread =
// channel). Emits global_prefetch_b8 on the streaming channel loop.
// ---------------------------------------------------------------------------
__global__ void k1_fea_cm(const float* __restrict__ x, float* __restrict__ ws) {
  const int patch = blockIdx.x;            // 0..1023
  const int b  = patch >> 6;
  const int l  = patch & 63;
  const int lh = l >> 3, lw = l & 7;
  const int tid = threadIdx.x;             // 256 threads
  const int i = tid >> 4, j = tid & 15;
  const size_t base = (size_t)b * CC * PLANE + (size_t)(lh * PP) * WW + (size_t)(lw * PP);

  __shared__ float sfea[PP][PP];

  // phase A: fea_raw[i][j] = mean_c x^2 ; consecutive lanes -> consecutive j (64B segments)
  float acc = 0.f;
  const float* px = x + base + (size_t)i * WW + j;
  for (int c = 0; c < CC; ++c) {
    __builtin_prefetch(px + (size_t)(c + 8) * PLANE, 0, 0);  // -> global_prefetch_b8
    float v = px[(size_t)c * PLANE];
    acc = fmaf(v, v, acc);
  }
  float fraw = acc * (1.f / 256.f);
  sfea[i][j] = fraw;
  __syncthreads();

  // fused 3x3 maxpool (stride 1, pad 1) then fea = fea_raw + mp
  float mp = -INFINITY;
  for (int dy = -1; dy <= 1; ++dy)
    for (int dx = -1; dx <= 1; ++dx) {
      int ii = i + dy, jj = j + dx;
      if (ii >= 0 && ii < PP && jj >= 0 && jj < PP) mp = fmaxf(mp, sfea[ii][jj]);
    }
  ws[OFF_FEA + patch * P2 + tid] = fraw + mp;

  // phase B: cm[c] = mean_{i,j} x^2 ; thread == channel, rows re-served from L2
  {
    const int c = tid;
    const float* pc = x + base + (size_t)c * PLANE;
    float a2 = 0.f;
    for (int r = 0; r < PP; ++r) {
      const float4* row = (const float4*)(pc + (size_t)r * WW);  // 64B-aligned (lw*16 floats)
      for (int q = 0; q < 4; ++q) {
        float4 v = row[q];
        a2 = fmaf(v.x, v.x, fmaf(v.y, v.y, fmaf(v.z, v.z, fmaf(v.w, v.w, a2))));
      }
    }
    float cmv = a2 * (1.f / 256.f);
    ws[OFF_CM + patch * P2 + c] = cmv;
    ((_Float16*)(ws + OFF_CMH))[patch * P2 + c] = (_Float16)cmv;
  }
}

// ---------------------------------------------------------------------------
// K2: 7x7 conv1 on fea. The 22x22 zero-padded LDS tile interior is filled by
// the Tensor Data Mover (row-padded DMA); BN statistics go to per-block
// partials (deterministic, no float atomics).
// ---------------------------------------------------------------------------
__global__ void k2_conv1_stats(const float* __restrict__ w1, const float* __restrict__ b1,
                               float* __restrict__ ws) {
  const int patch = blockIdx.x;
  const int tid = threadIdx.x;
  const int i = tid >> 4, j = tid & 15;
  __shared__ float tile[PP + 6][PP + 6];
  __shared__ float wk[49];
  __shared__ float rs[256], rss[256];

  if (tid < 49) wk[tid] = w1[tid];
  for (int idx = tid; idx < (PP + 6) * (PP + 6); idx += 256) (&tile[0][0])[idx] = 0.f;
  __syncthreads();

  // TDM fills tile[3..18][3..18] directly from ws (one wave issues the DMA)
  if (tid < 32) {
    tdm_load_tile16x16_pad22(ws + OFF_FEA + patch * P2, &tile[3][3]);
    __builtin_amdgcn_s_wait_tensorcnt(0);
  }
  __syncthreads();

  float acc = b1[0];
  for (int dy = 0; dy < 7; ++dy)
    for (int dx = 0; dx < 7; ++dx)
      acc = fmaf(tile[i + dy][j + dx], wk[dy * 7 + dx], acc);
  ws[OFF_HPRE + patch * P2 + tid] = acc;

  rs[tid] = acc; rss[tid] = acc * acc;
  __syncthreads();
  for (int s = 128; s > 0; s >>= 1) {
    if (tid < s) { rs[tid] += rs[tid + s]; rss[tid] += rss[tid + s]; }
    __syncthreads();
  }
  if (tid == 0) {
    ws[OFF_PSUM + patch] = rs[0];
    ws[OFF_PSSQ + patch] = rss[0];
  }
}

// K2b: deterministic reduction of the 1024 per-block BN partials.
__global__ void k2b_reduce_stats(float* __restrict__ ws) {
  const int tid = threadIdx.x;
  __shared__ float rs[256], rss[256];
  float s = 0.f, ss = 0.f;
  for (int q = tid; q < NP; q += 256) { s += ws[OFF_PSUM + q]; ss += ws[OFF_PSSQ + q]; }
  rs[tid] = s; rss[tid] = ss;
  __syncthreads();
  for (int st = 128; st > 0; st >>= 1) {
    if (tid < st) { rs[tid] += rs[tid + st]; rss[tid] += rss[tid + st]; }
    __syncthreads();
  }
  if (tid == 0) { ws[OFF_STAT + 0] = rs[0]; ws[OFF_STAT + 1] = rss[0]; }
}

// ---------------------------------------------------------------------------
// K3: BN + ReLU -> conv2 (7x7) -> sigmoid -> spatial softmax -> spatial score.
// Entirely fused per patch; softmax via LDS tree reductions.
// ---------------------------------------------------------------------------
__global__ void k3_bn_conv2_softmax(const float* __restrict__ g, const float* __restrict__ be,
                                    const float* __restrict__ w2, const float* __restrict__ b2,
                                    float* __restrict__ ws) {
  const int patch = blockIdx.x;
  const int tid = threadIdx.x;
  const int i = tid >> 4, j = tid & 15;
  __shared__ float tile[PP + 6][PP + 6];
  __shared__ float wk[49];
  __shared__ float red[256];

  if (tid < 49) wk[tid] = w2[tid];
  for (int idx = tid; idx < (PP + 6) * (PP + 6); idx += 256) (&tile[0][0])[idx] = 0.f;

  const float N = (float)(NP * P2);
  const float mu = ws[OFF_STAT + 0] / N;
  const float var = ws[OFF_STAT + 1] / N - mu * mu;
  const float istd = rsqrtf(var + 1e-5f);
  const float gamma = g[0], beta = be[0];
  __syncthreads();

  float hp = ws[OFF_HPRE + patch * P2 + tid];
  float hv = fmaxf((hp - mu) * istd * gamma + beta, 0.f);
  tile[i + 3][j + 3] = hv;
  __syncthreads();

  float acc = b2[0];
  for (int dy = 0; dy < 7; ++dy)
    for (int dx = 0; dx < 7; ++dx)
      acc = fmaf(tile[i + dy][j + dx], wk[dy * 7 + dx], acc);
  float sw = 1.f / (1.f + __expf(-acc));
  float fea = ws[OFF_FEA + patch * P2 + tid];
  float s = fea * sw * INV_TEMP;

  // block softmax over 256 spatial logits
  red[tid] = s; __syncthreads();
  for (int st = 128; st > 0; st >>= 1) { if (tid < st) red[tid] = fmaxf(red[tid], red[tid + st]); __syncthreads(); }
  float mx = red[0]; __syncthreads();
  float e = __expf(s - mx);
  red[tid] = e; __syncthreads();
  for (int st = 128; st > 0; st >>= 1) { if (tid < st) red[tid] += red[tid + st]; __syncthreads(); }
  float S = red[0]; __syncthreads();
  float p = e / S;                         // spatial_score = sum_k softmax_k
  red[tid] = p; __syncthreads();
  for (int st = 128; st > 0; st >>= 1) { if (tid < st) red[tid] += red[tid + st]; __syncthreads(); }
  if (tid == 0) ws[OFF_SS + patch] = red[0];
}

// ---------------------------------------------------------------------------
// K4: channel attention. Both GEMMs on the CDNA5 matrix cores:
//   hidden(16x16) = cm(16x256) @ lin1(256x16)   -> K split across 8 waves,
//                                                  1 v_wmma_f32_16x16x32_f16 each,
//                                                  LDS partial reduction + bias + relu
//   cw(16x256)    = hidden(16x32,K-padded) @ lin2 -> 2 N-tiles per wave
// Fragment lane layouts use contiguous K order for A and B (16-bit A: lane<16 holds M=lane,
// halves h -> K = h (+16 for lanes 16..31); B: lane<16 N=lane K=h,
// lanes 16..31 K=h+16; D: vgpr r -> M = r (+8 for lanes 16..31), N = lane&15).
// Fused sigmoid + channel softmax + channel score, all in LDS.
// ---------------------------------------------------------------------------
__global__ void k4_channel_att(const float* __restrict__ l1w, const float* __restrict__ l1b,
                               const float* __restrict__ l2w, const float* __restrict__ l2b,
                               float* __restrict__ ws) {
  const int tid = threadIdx.x;             // 256 threads = 8 waves
  const int wv = tid >> 5, lane = tid & 31;
  const int m0 = blockIdx.x * 16;          // 64 blocks cover M = 1024
  const _Float16* cmh = (const _Float16*)(ws + OFF_CMH);

  __shared__ _Float16 l1h[256 * 16];       // lin1 weights as f16, row-major [k][n]
  __shared__ float    part[8][256];        // per-wave GEMM1 partials
  __shared__ _Float16 hidA[16][32];        // GEMM2 A tile (K padded 16->32)
  __shared__ float    scw[16][256];        // cw then channel logits
  __shared__ float    red[16][17];
  __shared__ float    rowmax[16], rowsum[16];

  for (int idx = tid; idx < 4096; idx += 256) l1h[idx] = (_Float16)l1w[idx];
  __syncthreads();

  // ---- GEMM1: wave wv handles K-chunk [wv*32, wv*32+32)
  {
    const int kb = wv * 32;
    const int M = lane & 15;
    const int koffA = (lane < 16) ? 0 : 16;
    v16h a, bf;
    for (int h = 0; h < 16; ++h) {
      int K = h + koffA;
      a[h] = cmh[(m0 + M) * 256 + kb + K];
    }
    const int koffB = (lane < 16) ? 0 : 16;
    for (int h = 0; h < 16; ++h)
      bf[h] = l1h[(kb + koffB + h) * 16 + (lane & 15)];
    v8f c = {};
    c = __builtin_amdgcn_wmma_f32_16x16x32_f16(false, a, false, bf, (short)0, c, false, false);
    const int moff = (lane < 16) ? 0 : 8;
    for (int r = 0; r < 8; ++r) part[wv][(r + moff) * 16 + (lane & 15)] = c[r];
  }
  __syncthreads();

  // reduce K-partials, bias, relu, stage f16 A tile for GEMM2
  {
    const int mm = tid >> 4, nn = tid & 15;
    float hsum = 0.f;
    for (int w = 0; w < 8; ++w) hsum += part[w][tid];
    hsum = fmaxf(hsum + l1b[nn], 0.f);
    hidA[mm][nn] = (_Float16)hsum;
    hidA[mm][nn + 16] = (_Float16)0.f;     // zero-pad K 16..31
  }
  __syncthreads();

  // ---- GEMM2: wave wv computes N-tiles 2*wv and 2*wv+1
  {
    const int M = lane & 15;
    const int koffA = (lane < 16) ? 0 : 16;
    v16h a;
    for (int h = 0; h < 16; ++h) {
      int K = h + koffA;
      a[h] = hidA[M][K];
    }
    for (int t = 0; t < 2; ++t) {
      const int nt = wv * 2 + t;
      v16h bf;
      if (lane < 16) { for (int h = 0; h < 16; ++h) bf[h] = (_Float16)l2w[h * 256 + nt * 16 + lane]; }
      else           { for (int h = 0; h < 16; ++h) bf[h] = (_Float16)0.f; }  // K>=16 padding
      v8f c = {};
      c = __builtin_amdgcn_wmma_f32_16x16x32_f16(false, a, false, bf, (short)0, c, false, false);
      const int moff = (lane < 16) ? 0 : 8;
      const int n = nt * 16 + (lane & 15);
      for (int r = 0; r < 8; ++r) {
        float z = c[r] + l2b[n];
        scw[r + moff][n] = 1.f / (1.f + __expf(-z));   // cw = sigmoid(.)
      }
    }
  }
  __syncthreads();

  // ---- channel softmax score per patch row: 16 threads per row, 16 cols each
  {
    const int mm = tid >> 4, sub = tid & 15;
    const int p = m0 + mm;
    float lm = -INFINITY;
    for (int q = 0; q < 16; ++q) {
      int cch = sub * 16 + q;
      float sv = ws[OFF_CM + p * 256 + cch] * scw[mm][cch] * INV_TEMP;
      scw[mm][cch] = sv;
      lm = fmaxf(lm, sv);
    }
    red[mm][sub] = lm; __syncthreads();
    if (sub == 0) { float r0 = red[mm][0]; for (int q = 1; q < 16; ++q) r0 = fmaxf(r0, red[mm][q]); rowmax[mm] = r0; }
    __syncthreads();
    float ls = 0.f;
    for (int q = 0; q < 16; ++q) ls += __expf(scw[mm][sub * 16 + q] - rowmax[mm]);
    red[mm][sub] = ls; __syncthreads();
    if (sub == 0) { float r0 = 0.f; for (int q = 0; q < 16; ++q) r0 += red[mm][q]; rowsum[mm] = r0; }
    __syncthreads();
    float inv = 1.f / rowsum[mm];
    float sc = 0.f;                         // channel_score = sum_c softmax_c
    for (int q = 0; q < 16; ++q) sc += __expf(scw[mm][sub * 16 + q] - rowmax[mm]) * inv;
    red[mm][sub] = sc; __syncthreads();
    if (sub == 0) { float r0 = 0.f; for (int q = 0; q < 16; ++q) r0 += red[mm][q]; ws[OFF_CS + p] = r0; }
  }
}

// ---------------------------------------------------------------------------
// K5: per-batch stable bottom-16 selection (matches lax.top_k(-scores, 16)
// tie-breaking: equal scores pick lower index) and mask broadcast to 128x128.
// ---------------------------------------------------------------------------
__global__ void k5_mask(float* __restrict__ out, const float* __restrict__ ws) {
  const int b = blockIdx.x, tid = threadIdx.x;
  __shared__ float sc[LL];
  __shared__ int   sel[LL];

  if (tid < LL) {
    int p = b * LL + tid;
    sc[tid] = ALPHA_ * ws[OFF_SS + p] + (1.f - ALPHA_) * ws[OFF_CS + p];
  }
  __syncthreads();
  if (tid < LL) {
    float sv = sc[tid]; int cnt = 0;
    for (int q = 0; q < LL; ++q)
      cnt += (sc[q] < sv) || (sc[q] == sv && q < tid);
    sel[tid] = (cnt < TOPK) ? 1 : 0;       // 1 = masked out (zero)
  }
  __syncthreads();
  float* ob = out + (size_t)b * PLANE;
  for (int pix = tid; pix < PLANE; pix += 256) {
    int hh = pix >> 7, wwp = pix & 127;
    ob[pix] = sel[(hh >> 4) * 8 + (wwp >> 4)] ? 0.f : 1.f;
  }
}

// ---------------------------------------------------------------------------
extern "C" void kernel_launch(void* const* d_in, const int* in_sizes, int n_in,
                              void* d_out, int out_size, void* d_ws, size_t ws_size,
                              hipStream_t stream) {
  if (ws_size < WS_NEEDED_BYTES) return;
  const float* x   = (const float*)d_in[0];
  const float* w1  = (const float*)d_in[1];
  const float* b1  = (const float*)d_in[2];
  const float* g   = (const float*)d_in[3];
  const float* be  = (const float*)d_in[4];
  const float* w2  = (const float*)d_in[5];
  const float* b2  = (const float*)d_in[6];
  const float* l1w = (const float*)d_in[7];
  const float* l1b = (const float*)d_in[8];
  const float* l2w = (const float*)d_in[9];
  const float* l2b = (const float*)d_in[10];
  float* ws  = (float*)d_ws;
  float* out = (float*)d_out;

  hipLaunchKernelGGL(k1_fea_cm,           dim3(NP), dim3(256), 0, stream, x, ws);
  hipLaunchKernelGGL(k2_conv1_stats,      dim3(NP), dim3(256), 0, stream, w1, b1, ws);
  hipLaunchKernelGGL(k2b_reduce_stats,    dim3(1),  dim3(256), 0, stream, ws);
  hipLaunchKernelGGL(k3_bn_conv2_softmax, dim3(NP), dim3(256), 0, stream, g, be, w2, b2, ws);
  hipLaunchKernelGGL(k4_channel_att,      dim3(64), dim3(256), 0, stream, l1w, l1b, l2w, l2b, ws);
  hipLaunchKernelGGL(k5_mask,             dim3(BB), dim3(256), 0, stream, out, ws);
}
